// Attention_77799037600428
// MI455X (gfx1250) — compile-verified, matrix operand order checked by reading, not run on hardware
//
#include <hip/hip_runtime.h>
#include <hip/hip_bf16.h>
#include <stdint.h>

typedef __attribute__((ext_vector_type(16))) __bf16 v16bf;
typedef __attribute__((ext_vector_type(8)))  float  v8f;

#define Bsz  2
#define Nseq 2048
#define Dmod 1024
#define Hn   16
#define DHd  64
#define QK_SCALE 0.125f   // 64^-0.5

// round-to-nearest-even fp32 -> bf16 bits
__device__ __forceinline__ uint16_t f2bf(float f) {
  uint32_t u = __float_as_uint(f);
  u += 0x7FFFu + ((u >> 16) & 1u);
  return (uint16_t)(u >> 16);
}

// 16-bit A/B fragment: per lane two contiguous 8-element (16B) chunks at
// base (= ptr + k0 + half*8) and base+16 -> two b128 loads (global or ds)
__device__ __forceinline__ void load_frag16(v16bf& fr, const uint16_t* base) {
  ((uint4*)&fr)[0] = *(const uint4*)(base);
  ((uint4*)&fr)[1] = *(const uint4*)(base + 16);
}

// ---------------- Tensor Data Mover staging of B tiles ----------------
#if defined(__has_builtin)
#if __has_builtin(__builtin_amdgcn_tensor_load_to_lds)
#define HAVE_TDM 1
#endif
#endif

#ifdef HAVE_TDM
typedef __attribute__((ext_vector_type(4))) unsigned int u32x4;
typedef __attribute__((ext_vector_type(8))) int          i32x8;
typedef __attribute__((ext_vector_type(4))) int          i32x4;

// DMA a 64-row x 32-halfword tile of a [nrows x 1024] bf16 matrix (row-major,
// contiguous K) into LDS at lds_off. Tile rows land contiguously: LDS[r*32+k].
__device__ __forceinline__ void tdm_load_b_tile(const uint16_t* tile_start,
                                                int nrows_total, uint32_t lds_off) {
  uint64_t ga = (uint64_t)(uintptr_t)tile_start;       // byte address of tile start
  u32x4 g0;
  g0[0] = 1u;                                          // count=1, user descriptor
  g0[1] = lds_off;                                     // lds_addr [63:32]
  g0[2] = (uint32_t)ga;                                // global_addr[31:0]
  g0[3] = (uint32_t)((ga >> 32) & 0x01FFFFFFu)         // global_addr[56:32]
        | (2u << 30);                                  // type=2 (image)
  i32x8 g1;
  g1[0] = (int)0x00010000u;                            // data_size=1 (2B); mask=0
  g1[1] = (int)((1024u & 0xFFFFu) << 16);              // tensor_dim0=1024 (lo16 @63:48)
  g1[2] = (int)(((uint32_t)nrows_total & 0xFFFFu) << 16); // dim0 hi=0 | tensor_dim1 lo16
  g1[3] = (int)((((uint32_t)nrows_total >> 16) & 0xFFFFu)
        | (32u << 16));                                // tensor_dim1 hi | tile_dim0=32
  g1[4] = (int)64;                                     // tile_dim1=64, tile_dim2=0
  g1[5] = (int)1024;                                   // tensor_dim0_stride lo32
  g1[6] = 0;                                           // stride0 hi / stride1 lo
  g1[7] = 0;
  i32x4 z4 = {};
  i32x8 z8 = {};
  __builtin_amdgcn_tensor_load_to_lds(g0, g1, z4, z4, z8, 0);
}
#else
// Fallback: 256 threads cooperatively copy the 64x32 tile (8 bf16 each).
__device__ __forceinline__ void coop_load_b_tile(uint16_t* dst, const uint16_t* wT,
                                                 int row0, int kk) {
  int e = threadIdx.x * 8;
  int r = e >> 5, c = e & 31;
  *(uint4*)(dst + e) = *(const uint4*)(wT + (size_t)(row0 + r) * 1024 + kk + c);
}
#endif

// ---------------- conversion / transpose kernels ----------------
__global__ __launch_bounds__(256) void k_cvt_x(const float* __restrict__ x,
                                               uint16_t* __restrict__ xb, int n) {
  int i = blockIdx.x * 256 + threadIdx.x;
  if (i < n) xb[i] = f2bf(x[i]);
}

__global__ __launch_bounds__(256) void k_transpose(const float* __restrict__ w,
                                                   uint16_t* __restrict__ wT,
                                                   int rows /*k*/, int cols /*n*/) {
  int i = blockIdx.x * 256 + threadIdx.x;
  if (i < rows * cols) {
    int k = i / cols, n = i % cols;
    wT[n * rows + k] = f2bf(w[i]);   // wT is [n][k]: contiguous K per output col
  }
}

// ============ block-cooperative GEMM core (M=128/block, N=64/block) ============
// Each of the 8 waves owns a 16-row strip; the 64x32 B tile is shared through
// LDS, TDM double-buffered; A fragments are software-pipelined from global.
template <int NROWS_B>
__device__ __forceinline__ void gemm_tile_128x64(
    const uint16_t* __restrict__ A, const uint16_t* __restrict__ wT,
    int rowA0, int colB0, v8f (&acc)[4], uint16_t (&ldsB)[2][64 * 32]) {
  const int lane = threadIdx.x & 31;
  const int half = lane >> 4, ln = lane & 15;
  const uint16_t* arow = A + (size_t)(rowA0 + ln) * Dmod;

  // preload B tile 0
#ifdef HAVE_TDM
  if (threadIdx.x < 32) {
    tdm_load_b_tile(wT + (size_t)colB0 * Dmod, NROWS_B,
                    (uint32_t)(uintptr_t)&ldsB[0][0]);
    __builtin_amdgcn_s_wait_tensorcnt(0);
  }
#else
  coop_load_b_tile(&ldsB[0][0], wT, colB0, 0);
#endif
  __syncthreads();

  v16bf aCur; load_frag16(aCur, arow + half * 8);

  for (int i = 0; i < Dmod / 32; ++i) {
    const int kk = i * 32;
    // stage next B tile into the alternate buffer (no wait yet)
    if (i + 1 < Dmod / 32) {
#ifdef HAVE_TDM
      if (threadIdx.x < 32)
        tdm_load_b_tile(wT + (size_t)colB0 * Dmod + kk + 32, NROWS_B,
                        (uint32_t)(uintptr_t)&ldsB[(i + 1) & 1][0]);
#else
      coop_load_b_tile(&ldsB[(i + 1) & 1][0], wT, colB0, kk + 32);
#endif
    }
    // pipeline next A fragment ahead of this step's WMMAs
    v16bf aNext = aCur;
    if (i + 1 < Dmod / 32) load_frag16(aNext, arow + kk + 32 + half * 8);
    __builtin_prefetch(arow + kk + 256, 0, 3);

    const uint16_t* bbase = &ldsB[i & 1][0];
#pragma unroll
    for (int t = 0; t < 4; ++t) {
      v16bf b;
      const uint16_t* pb = bbase + (t * 16 + ln) * 32 + half * 8;
      ((uint4*)&b)[0] = *(const uint4*)(pb);
      ((uint4*)&b)[1] = *(const uint4*)(pb + 16);
      acc[t] = __builtin_amdgcn_wmma_f32_16x16x32_bf16(
          false, aCur, false, b, (short)0, acc[t], false, false);
    }
#ifdef HAVE_TDM
    if (threadIdx.x < 32) __builtin_amdgcn_s_wait_tensorcnt(0);
#endif
    __syncthreads();          // tile i reads done; tile i+1 resident
    aCur = aNext;
  }
}

// ---------------- fused QKV projection GEMM ----------------
// C[4096 x 3072] = xb * w_qkvT. Block: 128 rows x 64 cols.
// Epilogue scatters to Q(b,h,n,dh)*scale, K(b,h,n,dh), VT(b,h,dh,n) in bf16.
__global__ __launch_bounds__(256) void k_gemm_qkv(const uint16_t* __restrict__ xb,
                                                  const uint16_t* __restrict__ wT,
                                                  uint16_t* __restrict__ Q,
                                                  uint16_t* __restrict__ K,
                                                  uint16_t* __restrict__ VT) {
  __shared__ uint16_t ldsB[2][64 * 32];
  const int lane = threadIdx.x & 31;
  const int widx = threadIdx.x >> 5;
  const int rb   = blockIdx.x / 48;          // 32 row blocks of 128
  const int cg   = blockIdx.x % 48;          // 48 col groups of 64
  const int rt   = rb * 8 + widx;            // this wave's 16-row tile
  const int half = lane >> 4, ln = lane & 15;

  v8f acc[4] = {};
  gemm_tile_128x64<3072>(xb, wT, rt * 16, cg * 64, acc, ldsB);

#pragma unroll
  for (int t = 0; t < 4; ++t)
#pragma unroll
    for (int r = 0; r < 8; ++r) {
      int row = rt * 16 + r + 8 * half;      // C layout: M = r + 8*half
      int c   = cg * 64 + t * 16 + ln;       // N = lane&15
      int qkv = c >> 10, rem = c & 1023;
      int h = rem >> 6, dh = rem & 63;
      int b_ = row >> 11, n_ = row & 2047;
      float v = acc[t][r];
      if (qkv == 0)      Q [(((b_ * Hn + h) * Nseq) + n_) * DHd + dh] = f2bf(v * QK_SCALE);
      else if (qkv == 1) K [(((b_ * Hn + h) * Nseq) + n_) * DHd + dh] = f2bf(v);
      else               VT[(((b_ * Hn + h) * DHd) + dh) * Nseq + n_] = f2bf(v);
    }
}

// ---------------- causal flash attention ----------------
// One wave handles one (b,h) and 16 query rows. Online softmax, keys in
// blocks of 32 (two 16-wide S tiles). P re-laid out via per-wave LDS tile.
__global__ __launch_bounds__(256) void k_attn(const uint16_t* __restrict__ Q,
                                              const uint16_t* __restrict__ K,
                                              const uint16_t* __restrict__ VT,
                                              uint16_t* __restrict__ O) {
  __shared__ uint16_t ldsP[8 * 16 * 32];          // 1KB per wave
  const int lane = threadIdx.x & 31;
  const int widx = threadIdx.x >> 5;
  const int w    = blockIdx.x * 8 + widx;
  const int bh   = w >> 7;                         // / 128 query tiles
  const int qt   = w & 127;
  const int b_   = bh >> 4, h = bh & 15;
  const int half = lane >> 4, ln = lane & 15;
  const int qi0  = qt * 16;
  uint16_t* P = ldsP + widx * 512;

  // Q A-fragments (dh 0..31 and 32..63), resident for the whole key loop
  const uint16_t* qbase = Q + ((b_ * Hn + h) * Nseq + (qi0 + ln)) * DHd;
  v16bf aq0, aq1;
  load_frag16(aq0, qbase + half * 8);
  load_frag16(aq1, qbase + 32 + half * 8);

  v8f o[4] = {};
  float mrow[8], lrow[8];
#pragma unroll
  for (int r = 0; r < 8; ++r) { mrow[r] = -1e30f; lrow[r] = 0.f; }

  const int nkb = (qi0 + 16 + 31) >> 5;            // causal key-block bound
  for (int kb = 0; kb < nkb; ++kb) {
    const int j0 = kb * 32;
    const uint16_t* kb0 = K + ((b_ * Hn + h) * Nseq + (j0 + ln)) * DHd;
    const uint16_t* kb1 = kb0 + 16 * DHd;

    v8f s0 = {}, s1 = {};
    { v16bf bt; load_frag16(bt, kb0 + half * 8);
      s0 = __builtin_amdgcn_wmma_f32_16x16x32_bf16(false, aq0, false, bt, (short)0, s0, false, false); }
    { v16bf bt; load_frag16(bt, kb0 + 32 + half * 8);
      s0 = __builtin_amdgcn_wmma_f32_16x16x32_bf16(false, aq1, false, bt, (short)0, s0, false, false); }
    { v16bf bt; load_frag16(bt, kb1 + half * 8);
      s1 = __builtin_amdgcn_wmma_f32_16x16x32_bf16(false, aq0, false, bt, (short)0, s1, false, false); }
    { v16bf bt; load_frag16(bt, kb1 + 32 + half * 8);
      s1 = __builtin_amdgcn_wmma_f32_16x16x32_bf16(false, aq1, false, bt, (short)0, s1, false, false); }

    // causal mask + online softmax; width-16 reductions (wave32 halves)
#pragma unroll
    for (int r = 0; r < 8; ++r) {
      int qi = qi0 + r + 8 * half;
      float v0 = ((j0 + ln)      <= qi) ? s0[r] : -1e30f;
      float v1 = ((j0 + 16 + ln) <= qi) ? s1[r] : -1e30f;
      float mr = fmaxf(v0, v1);
#pragma unroll
      for (int d = 1; d < 16; d <<= 1) mr = fmaxf(mr, __shfl_xor(mr, d, 16));
      float mnew  = fmaxf(mrow[r], mr);
      float alpha = __expf(mrow[r] - mnew);
      float p0 = __expf(v0 - mnew);
      float p1 = __expf(v1 - mnew);
      float ps = p0 + p1;
#pragma unroll
      for (int d = 1; d < 16; d <<= 1) ps += __shfl_xor(ps, d, 16);
      lrow[r] = lrow[r] * alpha + ps;
      mrow[r] = mnew;
      o[0][r] *= alpha; o[1][r] *= alpha; o[2][r] *= alpha; o[3][r] *= alpha;
      int m = r + 8 * half;
      P[m * 32 + ln]      = f2bf(p0);
      P[m * 32 + 16 + ln] = f2bf(p1);
    }

    // reload P in A-fragment layout (LDS in-order within a wave)
    v16bf ap;
    {
      const uint16_t* pb = P + ln * 32 + half * 8;
      ((uint4*)&ap)[0] = *(const uint4*)(pb);
      ((uint4*)&ap)[1] = *(const uint4*)(pb + 16);
    }

    // PV: B-fragments from VT (contiguous along key axis)
#pragma unroll
    for (int t = 0; t < 4; ++t) {
      v16bf bv;
      load_frag16(bv, VT + ((b_ * Hn + h) * DHd + t * 16 + ln) * Nseq + j0 + half * 8);
      o[t] = __builtin_amdgcn_wmma_f32_16x16x32_bf16(
          false, ap, false, bv, (short)0, o[t], false, false);
    }
  }

  // normalize + store attention output as bf16 [B*N, H*DH]
#pragma unroll
  for (int r = 0; r < 8; ++r) {
    float linv = 1.0f / lrow[r];
    int row = qi0 + r + 8 * half;
#pragma unroll
    for (int t = 0; t < 4; ++t) {
      int col = h * 64 + t * 16 + ln;
      O[(b_ * Nseq + row) * (Hn * DHd) + col] = f2bf(o[t][r] * linv);
    }
  }
}

// ---------------- output projection GEMM ----------------
// out[4096 x 1024] fp32 = Obf * w_outT. Block: 128 rows x 64 cols.
__global__ __launch_bounds__(256) void k_gemm_out(const uint16_t* __restrict__ Obf,
                                                  const uint16_t* __restrict__ wT,
                                                  float* __restrict__ out) {
  __shared__ uint16_t ldsB[2][64 * 32];
  const int lane = threadIdx.x & 31;
  const int widx = threadIdx.x >> 5;
  const int rb   = blockIdx.x >> 4;          // 32 row blocks of 128
  const int cg   = blockIdx.x & 15;          // 16 col groups of 64
  const int rt   = rb * 8 + widx;
  const int half = lane >> 4, ln = lane & 15;

  v8f acc[4] = {};
  gemm_tile_128x64<1024>(Obf, wT, rt * 16, cg * 64, acc, ldsB);

#pragma unroll
  for (int t = 0; t < 4; ++t)
#pragma unroll
    for (int r = 0; r < 8; ++r)
      out[(size_t)(rt * 16 + r + 8 * half) * 1024 + cg * 64 + t * 16 + ln] = acc[t][r];
}

// ---------------- launch ----------------
extern "C" void kernel_launch(void* const* d_in, const int* in_sizes, int n_in,
                              void* d_out, int out_size, void* d_ws, size_t ws_size,
                              hipStream_t stream) {
  const float* x     = (const float*)d_in[0];
  // d_in[1] = attn_mask: causal tril, implemented analytically in-kernel
  const float* w_qkv = (const float*)d_in[2];
  const float* w_out = (const float*)d_in[3];
  float* out = (float*)d_out;

  uint16_t* ws    = (uint16_t*)d_ws;
  uint16_t* xb    = ws;                                   // [4096,1024] bf16
  uint16_t* wqkvT = xb    + 4096 * 1024;                  // [3072,1024]
  uint16_t* woutT = wqkvT + 3072 * 1024;                  // [1024,1024]
  uint16_t* Q     = woutT + 1024 * 1024;                  // [B,H,N,DH]
  uint16_t* K     = Q     + Bsz * Hn * Nseq * DHd;
  uint16_t* VT    = K     + Bsz * Hn * Nseq * DHd;        // [B,H,DH,N]
  uint16_t* Obf   = VT    + Bsz * Hn * Nseq * DHd;        // [4096,1024]

  k_cvt_x    <<<(4096 * 1024) / 256, 256, 0, stream>>>(x, xb, 4096 * 1024);
  k_transpose<<<(1024 * 3072) / 256, 256, 0, stream>>>(w_qkv, wqkvT, 1024, 3072);
  k_transpose<<<(1024 * 1024) / 256, 256, 0, stream>>>(w_out, woutT, 1024, 1024);
  k_gemm_qkv <<<1536, 256, 0, stream>>>(xb, wqkvT, Q, K, VT);
  k_attn     <<<512,  256, 0, stream>>>(Q, K, VT, Obf);
  k_gemm_out <<<512,  256, 0, stream>>>(Obf, woutT, out);
}
